// EdgeClassifier_83210696393431
// MI455X (gfx1250) — matrix operand order repaired and checked by reading, not compiled, not run on hardware
//
#include <hip/hip_runtime.h>
#include <hip/hip_bf16.h>

typedef __attribute__((ext_vector_type(16))) __bf16 v16bf;
typedef __attribute__((ext_vector_type(8)))  __bf16 v8bf;
typedef __attribute__((ext_vector_type(8)))  float  v8f;

#define GNN_N 65536
#define GNN_E 262144
#define WPB   8      // waves per block in the fused edge kernel (256 threads)

// -------------------------------------------------------------------------
// Pack f32 weight matrix [K x Ncols] into bf16 WMMA B-fragments.
// Fragment (kf,nf): per lane, 16 contiguous bf16 covering
//   k = kf*32 + (lane>>4)*16 + e ,  n = nf*16 + (lane&15)
// so the edge kernel loads one fragment as a single 32B (v16bf) load.
// -------------------------------------------------------------------------
__global__ void pack_b_kernel(const float* __restrict__ W, __bf16* __restrict__ out,
                              int Kact, int Ncols, int NF, int totalFrag)
{
    int t = blockIdx.x * blockDim.x + threadIdx.x;
    int f = t >> 5;
    if (f >= totalFrag) return;
    int lane = t & 31;
    int kf = f / NF, nf = f - kf * NF;
    int kb = kf * 32 + (lane >> 4) * 16;
    int n  = nf * 16 + (lane & 15);
#pragma unroll
    for (int e = 0; e < 16; ++e) {
        int k = kb + e;
        float v = (k < Kact) ? W[(size_t)k * Ncols + n] : 0.0f;
        out[(size_t)t * 16 + e] = (__bf16)v;
    }
}

// -------------------------------------------------------------------------
// Degree counts (scatter-mean denominator); shared by both layers.
// -------------------------------------------------------------------------
__global__ void count_kernel(const int* __restrict__ ei, float* __restrict__ cnt, int E)
{
    int e = blockIdx.x * blockDim.x + threadIdx.x;
    if (e < E) atomicAdd(&cnt[ei[E + e]], 1.0f);
}

// -------------------------------------------------------------------------
// Fused NNConv edge pass. Weights staged in LDS once per block; each wave
// grid-strides over 16-edge tiles:
//   h  = relu(edge_attr @ w1 + b1)     [16,256]  (16 WMMAs, K pad 10->32)
//   We = h @ w2 + b2                   [16,16*oc] (nf2*8 WMMAs, A in VGPRs, B in LDS)
//   msg[e][o] = sum_i x[src[e]][i]*We[e][i][o]   (VALU, negligible)
//   atomicAdd into ssum[dst[e]][o]
// -------------------------------------------------------------------------
__global__ __launch_bounds__(32 * WPB)
void nnconv_edge_kernel(const float* __restrict__ x,      // [N,16] layer input
                        const int*   __restrict__ ei,     // [2,E]
                        const float* __restrict__ ea,     // [E,10]
                        const __bf16* __restrict__ w1p,   // 16 frags packed (16KB)
                        const float* __restrict__ b1,     // [256]
                        const __bf16* __restrict__ w2p,   // 8*nf2 frags packed
                        const float* __restrict__ b2,     // [nf2*16]
                        float*       __restrict__ ssum,   // [N, out_c]
                        int E, int nf2, int log_out, int tiles)
{
    // ---- static LDS: 16K (w1) + 128K (w2 max) + 8*8K (h) + 8*1K (xs) = 216KB ----
    __shared__ __align__(16) __bf16 w1L[16 * 512];
    __shared__ __align__(16) __bf16 w2L[128 * 512];
    __shared__ __align__(16) __bf16 h_lds[WPB][16 * 256];
    __shared__ __align__(16) float  xs_lds[WPB][16 * 16];

    // ---- stage packed weights into LDS (whole block cooperates) ----
    {
        const uint4* g1 = (const uint4*)w1p;
        uint4*       d1 = (uint4*)w1L;
        for (int i = threadIdx.x; i < 16 * 512 * 2 / 16; i += blockDim.x) d1[i] = g1[i];
        const uint4* g2 = (const uint4*)w2p;
        uint4*       d2 = (uint4*)w2L;
        int nvec = nf2 * 8 * 512 * 2 / 16;   // frags * 1KB / 16B
        for (int i = threadIdx.x; i < nvec; i += blockDim.x) d2[i] = g2[i];
    }
    __syncthreads();

    const int wave   = threadIdx.x >> 5;
    const int lane   = threadIdx.x & 31;
    const int wid    = blockIdx.x * WPB + wave;
    const int nwaves = gridDim.x * WPB;

    const int row  = lane & 15;    // A row / C column-in-tile for this lane
    const int hi   = lane >> 4;    // lane half: selects K / M sub-ranges
    const int moff = hi * 8;       // C-layout: vgpr r holds row M = r + moff

    const v16bf* w1v = (const v16bf*)w1L;
    const v16bf* w2v = (const v16bf*)w2L;
    __bf16* hL = &h_lds[wave][0];
    float*  xL = &xs_lds[wave][0];

    const int out_mask = (1 << log_out) - 1;
    const int o = row & out_mask;              // lane-invariant across n-tiles

    for (int tile = wid; tile < tiles; tile += nwaves) {
        const int e0 = tile * 16;

        // prefetch next tile's edge_attr into cache
        {
            int en = e0 + nwaves * 16;
            if (en < E) __builtin_prefetch(&ea[(size_t)en * 10], 0, 0);
        }

        // ---- stage x[src] tile [16 edges][16 feats] into LDS ----
        {
            int esrc = ei[e0 + row];
#pragma unroll
            for (int c = 0; c < 8; ++c)
                xL[row * 16 + hi * 8 + c] = x[(size_t)esrc * 16 + hi * 8 + c];
        }

        // ---- build edge_attr A-fragment (K padded to 32, only K<10 live) ----
        v16bf afrag;
        {
            int kseg = hi * 16;
#pragma unroll
            for (int i = 0; i < 8; ++i) {
                int K = kseg + i;
                float v = (K < 10) ? ea[(size_t)(e0 + row) * 10 + K] : 0.0f;
                afrag[i]     = (__bf16)v;
                int K2 = kseg + 8 + i;
                float v2 = (K2 < 10) ? ea[(size_t)(e0 + row) * 10 + K2] : 0.0f;
                afrag[8 + i] = (__bf16)v2;
            }
        }

        // ---- MLP1: h = relu(ea @ w1 + b1), write bf16 h tile to LDS ----
#pragma unroll 1
        for (int nf = 0; nf < 16; ++nf) {
            v16bf bfrag = w1v[nf * 32 + lane];
            v8f acc = {};
            acc = __builtin_amdgcn_wmma_f32_16x16x32_bf16(
                      false, afrag, false, bfrag, (short)0, acc, false, false);
            int   ccol = nf * 16 + row;
            float bias = b1[ccol];
#pragma unroll
            for (int r = 0; r < 8; ++r) {
                float v = acc[r] + bias;
                v = v > 0.0f ? v : 0.0f;
                hL[(r + moff) * 256 + ccol] = (__bf16)v;
            }
        }
        asm volatile("s_wait_dscnt 0x0" ::: "memory");  // h + xs visible to this wave

        // ---- hoist all 8 h A-fragments into VGPRs (reused by every n-tile) ----
        v16bf hfrag[8];
#pragma unroll
        for (int kf = 0; kf < 8; ++kf) {
            int kbase = kf * 32 + hi * 16;
            v8bf lo  = *(const v8bf*)(hL + row * 256 + kbase);        // K = kbase..+7
            v8bf hi8 = *(const v8bf*)(hL + row * 256 + kbase + 8);    // K = kbase+8..+15
            hfrag[kf] = __builtin_shufflevector(lo, hi8,
                          0, 1, 2, 3, 4, 5, 6, 7, 8, 9, 10, 11, 12, 13, 14, 15);
        }

        // ---- MLP2 + fused per-edge matvec + scatter ----
        float msgacc[8];
        int   dsts[8];
#pragma unroll
        for (int r = 0; r < 8; ++r) { msgacc[r] = 0.0f; dsts[r] = ei[E + e0 + moff + r]; }

#pragma unroll 1
        for (int nf = 0; nf < nf2; ++nf) {
            v8f acc = {};
#pragma unroll
            for (int kf = 0; kf < 8; ++kf) {
                v16bf bfrag = w2v[(kf * nf2 + nf) * 32 + lane];
                acc = __builtin_amdgcn_wmma_f32_16x16x32_bf16(
                          false, hfrag[kf], false, bfrag, (short)0, acc, false, false);
            }
            int   c     = nf * 16 + row;        // flat We column
            int   i     = c >> log_out;         // input-feature index
            float bias2 = b2[c];
#pragma unroll
            for (int r = 0; r < 8; ++r) {
                float we = acc[r] + bias2;
                msgacc[r] += xL[(r + moff) * 16 + i] * we;
            }
        }

#pragma unroll
        for (int r = 0; r < 8; ++r)
            atomicAdd(&ssum[(size_t)dsts[r] * (out_mask + 1) + o], msgacc[r]);
    }
}

// -------------------------------------------------------------------------
// Node update: xout = relu(ssum/max(cnt,1) + xin @ root + bias)
// -------------------------------------------------------------------------
__global__ void node_update_kernel(const float* __restrict__ ssum,
                                   const float* __restrict__ cnt,
                                   const float* __restrict__ xin,
                                   const float* __restrict__ root,
                                   const float* __restrict__ bias,
                                   float* __restrict__ xout,
                                   int total, int in_c, int log_oc)
{
    int t = blockIdx.x * blockDim.x + threadIdx.x;
    if (t >= total) return;
    int n = t >> log_oc;
    int j = t & ((1 << log_oc) - 1);
    int oc = 1 << log_oc;
    float c = cnt[n]; c = c > 1.0f ? c : 1.0f;
    float s = ssum[t] / c + bias[j];
    const float* xr = xin + (size_t)n * in_c;
#pragma unroll 4
    for (int k = 0; k < in_c; ++k) s += xr[k] * root[k * oc + j];
    xout[t] = s > 0.0f ? s : 0.0f;
}

// -------------------------------------------------------------------------
// Edge classifier: sigmoid(concat(x2[src], x2[dst]) @ fc_w + fc_b)
// -------------------------------------------------------------------------
__global__ void classifier_kernel(const float* __restrict__ x2,
                                  const int*   __restrict__ ei,
                                  const float* __restrict__ fcw,
                                  const float* __restrict__ fcb,
                                  float* __restrict__ out, int E)
{
    int e = blockIdx.x * blockDim.x + threadIdx.x;
    if (e >= E) return;
    int s = ei[e], d = ei[E + e];
    float acc = fcb[0];
#pragma unroll
    for (int k = 0; k < 8; ++k) acc += x2[(size_t)s * 8 + k] * fcw[k];
#pragma unroll
    for (int k = 0; k < 8; ++k) acc += x2[(size_t)d * 8 + k] * fcw[8 + k];
    out[e] = 1.0f / (1.0f + __expf(-acc));
}

// -------------------------------------------------------------------------
extern "C" void kernel_launch(void* const* d_in, const int* in_sizes, int n_in,
                              void* d_out, int out_size, void* d_ws, size_t ws_size,
                              hipStream_t stream)
{
    const float* x      = (const float*)d_in[0];
    const int*   ei     = (const int*)  d_in[1];
    const float* ea     = (const float*)d_in[2];
    const float* w1_0   = (const float*)d_in[3];
    const float* b1_0   = (const float*)d_in[4];
    const float* w2_0   = (const float*)d_in[5];
    const float* b2_0   = (const float*)d_in[6];
    const float* root_0 = (const float*)d_in[7];
    const float* bias_0 = (const float*)d_in[8];
    const float* w1_1   = (const float*)d_in[9];
    const float* b1_1   = (const float*)d_in[10];
    const float* w2_1   = (const float*)d_in[11];
    const float* b2_1   = (const float*)d_in[12];
    const float* root_1 = (const float*)d_in[13];
    const float* bias_1 = (const float*)d_in[14];
    const float* fc_w   = (const float*)d_in[15];
    const float* fc_b   = (const float*)d_in[16];
    float* out = (float*)d_out;

    const int N = GNN_N, E = GNN_E;
    char* ws = (char*)d_ws;
    size_t off = 0;
    auto take = [&](size_t bytes) { size_t o = off; off = (off + bytes + 255) & ~(size_t)255; return o; };

    size_t off_cnt   = take((size_t)N * 4);
    size_t off_ssum0 = take((size_t)N * 16 * 4);
    size_t off_x1    = take((size_t)N * 16 * 4);
    size_t off_ssum1 = take((size_t)N * 8 * 4);
    size_t off_x2    = take((size_t)N * 8 * 4);
    size_t off_w1p0  = take((size_t)16  * 512 * 2);   // 1 kf * 16 nf frags
    size_t off_w2p0  = take((size_t)128 * 512 * 2);   // 8 kf * 16 nf frags
    size_t off_w1p1  = take((size_t)16  * 512 * 2);
    size_t off_w2p1  = take((size_t)64  * 512 * 2);   // 8 kf * 8 nf frags

    float*  cnt   = (float*)(ws + off_cnt);
    float*  ssum0 = (float*)(ws + off_ssum0);
    float*  x1    = (float*)(ws + off_x1);
    float*  ssum1 = (float*)(ws + off_ssum1);
    float*  x2    = (float*)(ws + off_x2);
    __bf16* w1p0  = (__bf16*)(ws + off_w1p0);
    __bf16* w2p0  = (__bf16*)(ws + off_w2p0);
    __bf16* w1p1  = (__bf16*)(ws + off_w1p1);
    __bf16* w2p1  = (__bf16*)(ws + off_w2p1);

    // zero accumulation buffers (re-done every call: deterministic)
    hipMemsetAsync(cnt,   0, (size_t)N * 4,      stream);
    hipMemsetAsync(ssum0, 0, (size_t)N * 16 * 4, stream);
    hipMemsetAsync(ssum1, 0, (size_t)N * 8 * 4,  stream);

    // pack weights to bf16 WMMA fragments
    pack_b_kernel<<<(16 * 32 + 255) / 256, 256, 0, stream>>>(w1_0, w1p0, 10, 256, 16, 16);
    pack_b_kernel<<<(128 * 32 + 255) / 256, 256, 0, stream>>>(w2_0, w2p0, 256, 256, 16, 128);
    pack_b_kernel<<<(16 * 32 + 255) / 256, 256, 0, stream>>>(w1_1, w1p1, 10, 256, 16, 16);
    pack_b_kernel<<<(64 * 32 + 255) / 256, 256, 0, stream>>>(w2_1, w2p1, 256, 128, 8, 64);

    // degree counts
    count_kernel<<<(E + 255) / 256, 256, 0, stream>>>(ei, cnt, E);

    const int tiles   = E / 16;            // 16384
    const int nblocks = 512;               // 8 waves each -> 4096 waves, 4 tiles/wave

    // ---- layer 0 ----
    nnconv_edge_kernel<<<nblocks, 32 * WPB, 0, stream>>>(
        x, ei, ea, w1p0, b1_0, w2p0, b2_0,
        ssum0, E, /*nf2=*/16, /*log_out=*/4, tiles);
    node_update_kernel<<<(N * 16 + 255) / 256, 256, 0, stream>>>(
        ssum0, cnt, x, root_0, bias_0, x1, N * 16, /*in_c=*/16, /*log_oc=*/4);

    // ---- layer 1 ----
    nnconv_edge_kernel<<<nblocks, 32 * WPB, 0, stream>>>(
        x1, ei, ea, w1p1, b1_1, w2p1, b2_1,
        ssum1, E, /*nf2=*/8, /*log_out=*/3, tiles);
    node_update_kernel<<<(N * 8 + 255) / 256, 256, 0, stream>>>(
        ssum1, cnt, x1, root_1, bias_1, x2, N * 8, /*in_c=*/16, /*log_oc=*/3);

    // ---- classifier ----
    classifier_kernel<<<(E + 255) / 256, 256, 0, stream>>>(x2, ei, fc_w, fc_b, out, E);
}
